// LSTMCell_15719580304274
// MI455X (gfx1250) — matrix operand order repaired and checked by reading, not compiled, not run on hardware
//
#include <hip/hip_runtime.h>
#include <math.h>

typedef __attribute__((ext_vector_type(16))) __bf16 v16bf;
typedef __attribute__((ext_vector_type(8)))  __bf16 v8bf;
typedef __attribute__((ext_vector_type(8)))  float  v8f;
typedef __attribute__((ext_vector_type(4)))  float  v4f;
typedef __attribute__((ext_vector_type(4)))  int    v4i;

#define HDIM 512
#define KTOT 1024          // D + H unified reduction dim
#define BM   128           // batch tile per block
#define BNG  32            // per-gate H tile per block
#define BK   64            // K step per stage (2 WMMA k-chunks)
#define NSTG (KTOT / BK)   // 16 stages (even -> clean 2x unroll)
#define LDA  72            // padded LDS row stride (bf16): 144B rows, 16B aligned
#define LDB  72

__device__ __forceinline__ float sigmoidf_fast(float x) {
    return 1.0f / (1.0f + __expf(-x));
}

__device__ __forceinline__ void cvt4_store(__bf16* d, v4f v) {
    d[0] = (__bf16)v[0]; d[1] = (__bf16)v[1];
    d[2] = (__bf16)v[2]; d[3] = (__bf16)v[3];
}

__device__ __forceinline__ void wait_async0() {
#if __has_builtin(__builtin_amdgcn_s_wait_asynccnt)
    __builtin_amdgcn_s_wait_asynccnt(0);
#else
    asm volatile("s_wait_asynccnt 0x0" ::: "memory");
#endif
}

// ---------------------------------------------------------------------------
// Pre-pass: f32 -> bf16, concatenating along K.
//   acts[b][0..511]  = data[b][:],  acts[b][512..1023] = hid[b][:]
//   Wcat[n][0..511]  = Wih[n][:],   Wcat[n][512..1023] = Whh[n][:]
// ---------------------------------------------------------------------------
__global__ __launch_bounds__(256)
void cvt_acts_kernel(const float* __restrict__ data,
                     const float* __restrict__ hid,
                     __bf16* __restrict__ acts) {
    const int t   = blockIdx.x * 256 + threadIdx.x;  // one float4 each
    const int row = t >> 8;                          // 256 float4 per 1024-row
    const int c4  = (t & 255) << 2;
    const float* src = (c4 < HDIM) ? (data + (size_t)row * HDIM + c4)
                                   : (hid  + (size_t)row * HDIM + (c4 - HDIM));
    cvt4_store(acts + (size_t)row * KTOT + c4, *(const v4f*)src);
}

__global__ __launch_bounds__(256)
void cvt_weights_kernel(const float* __restrict__ Wih,
                        const float* __restrict__ Whh,
                        __bf16* __restrict__ Wcat) {
    const int t   = blockIdx.x * 256 + threadIdx.x;
    const int row = t >> 8;                          // 0..2047
    const int c4  = (t & 255) << 2;
    const float* src = (c4 < HDIM) ? (Wih + (size_t)row * HDIM + c4)
                                   : (Whh + (size_t)row * HDIM + (c4 - HDIM));
    cvt4_store(Wcat + (size_t)row * KTOT + c4, *(const v4f*)src);
}

// ---------------------------------------------------------------------------
// Main GEMM: async-DMA bf16 tiles global->LDS (ASYNCcnt), pure ds+WMMA inner
// loop with compile-time-constant buffer indices (2x unrolled pipeline),
// fused LSTM epilogue. 8 waves = 4(M) x 2(N); each wave: 2 M-subtiles x
// 4 gates = 8 accumulators, 16 WMMAs/stage, 1 barrier/stage.
// ---------------------------------------------------------------------------
__global__ __launch_bounds__(256)
void lstm_gemm_async(const __bf16* __restrict__ acts,   // [B][1024]
                     const __bf16* __restrict__ Wcat,   // [2048][1024]
                     const float*  __restrict__ cur,
                     float* __restrict__ out_h,
                     float* __restrict__ out_c)
{
    __shared__ __bf16 Ash[2][BM * LDA];          // 128 x 64 bf16 (padded)
    __shared__ __bf16 Bsh[2][4 * BNG * LDB];     // 128 rows (4 gates x 32)

    const int tid  = threadIdx.x;
    const int lane = tid & 31;
    const int wave = tid >> 5;
    const int wm   = wave >> 1;         // 0..3
    const int wn   = wave & 1;          // 0..1

    const int bN = blockIdx.x * BNG;
    const int bM = blockIdx.y * BM;

    v8f acc[2][4] = {};

    // async-stage one BK tile (A: 512 x 16B chunks, B: 512 x 16B chunks)
    auto issue_stage = [&](int buf, int kb) {
        #pragma unroll
        for (int ii = 0; ii < 2; ++ii) {
            const int i   = tid + ii * 256;
            const int r   = i >> 2;              // 0..127
            const int c16 = (i & 3) << 4;        // 0,16,32,48 (bf16 elems)
            __builtin_amdgcn_global_load_async_to_lds_b128(
                (v4i*)(acts + (size_t)(bM + r) * KTOT + kb + c16),
                (v4i*)&Ash[buf][r * LDA + c16], 0, 0);
        }
        #pragma unroll
        for (int ii = 0; ii < 2; ++ii) {
            const int i   = tid + ii * 256;
            const int q   = i >> 2;              // 0..127 = g*32 + n
            const int c16 = (i & 3) << 4;
            const int g   = q >> 5;
            const int rr  = q & 31;
            __builtin_amdgcn_global_load_async_to_lds_b128(
                (v4i*)(Wcat + (size_t)(g * HDIM + bN + rr) * KTOT + kb + c16),
                (v4i*)&Bsh[buf][q * LDB + c16], 0, 0);
        }
    };

    // compute on a compile-time-constant buffer index
    auto compute = [&](int buf) {
        const int m  = lane & 15;
        const int kh = lane >> 4;
        #pragma unroll
        for (int kc = 0; kc < 2; ++kc) {
            v16bf af[2];
            #pragma unroll
            for (int mt = 0; mt < 2; ++mt) {
                const __bf16* ap =
                    &Ash[buf][(wm * 32 + mt * 16 + m) * LDA + kc * 32 + kh * 8];
                v8bf alo = *(const v8bf*)(ap);
                v8bf ahi = *(const v8bf*)(ap + 16);
                af[mt] = __builtin_shufflevector(alo, ahi,
                    0,1,2,3,4,5,6,7,8,9,10,11,12,13,14,15);
            }
            #pragma unroll
            for (int g = 0; g < 4; ++g) {
                const __bf16* bp =
                    &Bsh[buf][(g * BNG + wn * 16 + m) * LDB + kc * 32 + kh * 8];
                v8bf blo = *(const v8bf*)(bp);
                v8bf bhi = *(const v8bf*)(bp + 16);
                v16bf bfrag = __builtin_shufflevector(blo, bhi,
                    0,1,2,3,4,5,6,7,8,9,10,11,12,13,14,15);
                acc[0][g] = __builtin_amdgcn_wmma_f32_16x16x32_bf16(
                    false, af[0], false, bfrag, (short)0, acc[0][g], false, false);
                acc[1][g] = __builtin_amdgcn_wmma_f32_16x16x32_bf16(
                    false, af[1], false, bfrag, (short)0, acc[1][g], false, false);
            }
        }
    };

    // prologue
    issue_stage(0, 0);
    wait_async0();
    __syncthreads();

    // 2x-unrolled software pipeline: buffer index is a literal in each half
    for (int t = 0; t < NSTG; t += 2) {
        issue_stage(1, (t + 1) * BK);            // t+1 <= 15 always valid
        compute(0);
        wait_async0();
        __syncthreads();

        if (t + 2 < NSTG) issue_stage(0, (t + 2) * BK);
        compute(1);
        wait_async0();
        __syncthreads();
    }

    // fused LSTM epilogue
    const int gcol = bN + wn * 16 + (lane & 15);
    #pragma unroll
    for (int mt = 0; mt < 2; ++mt) {
        const int rbase = bM + wm * 32 + mt * 16 + ((lane >> 4) << 3);
        #pragma unroll
        for (int r = 0; r < 8; ++r) {
            const size_t idx = (size_t)(rbase + r) * HDIM + gcol;
            const float ig = sigmoidf_fast(acc[mt][0][r]);
            const float fg = sigmoidf_fast(acc[mt][1][r]);
            const float gt = tanhf(acc[mt][2][r]);
            const float og = sigmoidf_fast(acc[mt][3][r]);
            const float ns = fg * cur[idx] + ig * gt;
            out_h[idx] = og * tanhf(ns);
            out_c[idx] = ns;
        }
    }
}

// ---------------------------------------------------------------------------
// Fallback (no workspace): register-staged bf16 pipeline.
// ---------------------------------------------------------------------------
__global__ __launch_bounds__(256)
void lstm_cell_wmma_bf16(const float* __restrict__ data,
                         const float* __restrict__ hid,
                         const float* __restrict__ cur,
                         const float* __restrict__ Wih,
                         const float* __restrict__ Whh,
                         float* __restrict__ out_h,
                         float* __restrict__ out_c)
{
    __shared__ __bf16 Ash[2][BM * LDA];
    __shared__ __bf16 Bsh[2][4][BNG * LDB];

    const int tid  = threadIdx.x;
    const int lane = tid & 31;
    const int wave = tid >> 5;
    const int wm   = wave >> 1;
    const int wn   = wave & 1;

    const int bN = blockIdx.x * BNG;
    const int bM = blockIdx.y * BM;

    v8f acc[2][4] = {};
    v4f ar[8];
    v4f br[8];

    {
        #pragma unroll
        for (int ii = 0; ii < 8; ++ii) {
            const int i  = tid + ii * 256;
            ar[ii] = *(const v4f*)(data + (size_t)(bM + (i >> 4)) * HDIM + ((i & 15) << 2));
        }
        #pragma unroll
        for (int jj = 0; jj < 8; ++jj) {
            const int j  = tid + jj * 256;
            br[jj] = *(const v4f*)(Wih + (size_t)((j >> 9) * HDIM + bN + ((j >> 4) & 31)) * HDIM
                                       + ((j & 15) << 2));
        }
        #pragma unroll
        for (int ii = 0; ii < 8; ++ii) {
            const int i = tid + ii * 256;
            cvt4_store(&Ash[0][(i >> 4) * LDA + ((i & 15) << 2)], ar[ii]);
        }
        #pragma unroll
        for (int jj = 0; jj < 8; ++jj) {
            const int j = tid + jj * 256;
            cvt4_store(&Bsh[0][j >> 9][((j >> 4) & 31) * LDB + ((j & 15) << 2)], br[jj]);
        }
    }
    __syncthreads();

    int buf = 0;
    for (int t = 0; t < NSTG; ++t) {
        if (t + 1 < NSTG) {
            const int kb = (t + 1) * BK;
            const int kk = kb & (HDIM - 1);
            const float* Asrc = (kb < HDIM) ? data : hid;
            const float* Wsrc = (kb < HDIM) ? Wih  : Whh;
            #pragma unroll
            for (int ii = 0; ii < 8; ++ii) {
                const int i = tid + ii * 256;
                ar[ii] = *(const v4f*)(Asrc + (size_t)(bM + (i >> 4)) * HDIM + kk + ((i & 15) << 2));
            }
            #pragma unroll
            for (int jj = 0; jj < 8; ++jj) {
                const int j = tid + jj * 256;
                br[jj] = *(const v4f*)(Wsrc + (size_t)((j >> 9) * HDIM + bN + ((j >> 4) & 31)) * HDIM
                                            + kk + ((j & 15) << 2));
            }
        }

        const int m  = lane & 15;
        const int kh = lane >> 4;
        #pragma unroll
        for (int kc = 0; kc < 2; ++kc) {
            v16bf af[2];
            #pragma unroll
            for (int mt = 0; mt < 2; ++mt) {
                const __bf16* ap =
                    &Ash[buf][(wm * 32 + mt * 16 + m) * LDA + kc * 32 + kh * 8];
                v8bf alo = *(const v8bf*)(ap);
                v8bf ahi = *(const v8bf*)(ap + 16);
                af[mt] = __builtin_shufflevector(alo, ahi,
                    0,1,2,3,4,5,6,7,8,9,10,11,12,13,14,15);
            }
            #pragma unroll
            for (int g = 0; g < 4; ++g) {
                const __bf16* bp =
                    &Bsh[buf][g][(wn * 16 + m) * LDB + kc * 32 + kh * 8];
                v8bf blo = *(const v8bf*)(bp);
                v8bf bhi = *(const v8bf*)(bp + 16);
                v16bf bfrag = __builtin_shufflevector(blo, bhi,
                    0,1,2,3,4,5,6,7,8,9,10,11,12,13,14,15);
                acc[0][g] = __builtin_amdgcn_wmma_f32_16x16x32_bf16(
                    false, af[0], false, bfrag, (short)0, acc[0][g], false, false);
                acc[1][g] = __builtin_amdgcn_wmma_f32_16x16x32_bf16(
                    false, af[1], false, bfrag, (short)0, acc[1][g], false, false);
            }
        }

        if (t + 1 < NSTG) {
            #pragma unroll
            for (int ii = 0; ii < 8; ++ii) {
                const int i = tid + ii * 256;
                cvt4_store(&Ash[buf ^ 1][(i >> 4) * LDA + ((i & 15) << 2)], ar[ii]);
            }
            #pragma unroll
            for (int jj = 0; jj < 8; ++jj) {
                const int j = tid + jj * 256;
                cvt4_store(&Bsh[buf ^ 1][j >> 9][((j >> 4) & 31) * LDB + ((j & 15) << 2)], br[jj]);
            }
        }
        __syncthreads();
        buf ^= 1;
    }

    const int gcol = bN + wn * 16 + (lane & 15);
    #pragma unroll
    for (int mt = 0; mt < 2; ++mt) {
        const int rbase = bM + wm * 32 + mt * 16 + ((lane >> 4) << 3);
        #pragma unroll
        for (int r = 0; r < 8; ++r) {
            const size_t idx = (size_t)(rbase + r) * HDIM + gcol;
            const float ig = sigmoidf_fast(acc[mt][0][r]);
            const float fg = sigmoidf_fast(acc[mt][1][r]);
            const float gt = tanhf(acc[mt][2][r]);
            const float og = sigmoidf_fast(acc[mt][3][r]);
            const float ns = fg * cur[idx] + ig * gt;
            out_h[idx] = og * tanhf(ns);
            out_c[idx] = ns;
        }
    }
}

extern "C" void kernel_launch(void* const* d_in, const int* in_sizes, int n_in,
                              void* d_out, int out_size, void* d_ws, size_t ws_size,
                              hipStream_t stream) {
    const float* data = (const float*)d_in[0];   // [B, 512]
    const float* hid  = (const float*)d_in[1];   // [B, 512]
    const float* cur  = (const float*)d_in[2];   // [B, 512]
    const float* Wih  = (const float*)d_in[3];   // [2048, 512]
    const float* Whh  = (const float*)d_in[4];   // [2048, 512]

    const int B = in_sizes[0] / HDIM;            // 16384
    float* out_h = (float*)d_out;
    float* out_c = out_h + (size_t)B * HDIM;

    const size_t acts_elems = (size_t)B * KTOT;          // bf16
    const size_t w_elems    = (size_t)(4 * HDIM) * KTOT; // bf16
    const size_t need_bytes = (acts_elems + w_elems) * sizeof(__bf16);

    dim3 grid(HDIM / BNG, B / BM);               // (16, 128)

    if (ws_size >= need_bytes) {
        __bf16* acts = (__bf16*)d_ws;
        __bf16* Wcat = acts + acts_elems;
        cvt_acts_kernel<<<B, 256, 0, stream>>>(data, hid, acts);
        cvt_weights_kernel<<<4 * HDIM, 256, 0, stream>>>(Wih, Whh, Wcat);
        lstm_gemm_async<<<grid, 256, 0, stream>>>(acts, Wcat, cur, out_h, out_c);
    } else {
        lstm_cell_wmma_bf16<<<grid, 256, 0, stream>>>(data, hid, cur, Wih, Whh,
                                                      out_h, out_c);
    }
}
